// AuxiliaryGovernedAttention_19636590478145
// MI455X (gfx1250) — matrix operand order repaired and checked by reading, not compiled, not run on hardware
//
#include <hip/hip_runtime.h>

#define HID   4096
#define BOT   64
#define RDIM  48
#define NSLOT 1024
#define KSEL  8
#define VB    8
#define H4    1024
#define NTOK  4096

typedef __attribute__((ext_vector_type(16))) __bf16 v16bf;
typedef __attribute__((ext_vector_type(8)))  float  v8f;

union Frag16 {
    v16bf v;
    uint4 q[2];
    unsigned short u[16];
};

static __device__ __forceinline__ unsigned short f2bf(float f) {
    union { float f; unsigned u; } x{f};
    unsigned r = x.u + 0x7fffu + ((x.u >> 16) & 1u);   // round-to-nearest-even
    return (unsigned short)(r >> 16);
}

static __device__ __forceinline__ float gelu_tanh(float x) {
    float x3 = x * x * x;
    float t = tanhf(0.7978845608028654f * (x + 0.044715f * x3));
    return 0.5f * x * (1.0f + t);
}

// ---------------------------------------------------------------------------
// 0) zero accumulators (learned[NTOK], lv_sum[1])
// ---------------------------------------------------------------------------
__global__ void k_zero(float* learned, float* lv_sum) {
    int id = blockIdx.x * blockDim.x + threadIdx.x;
    if (id < NTOK) learned[id] = 0.0f;
    if (id == 0)   lv_sum[0] = 0.0f;
}

// ---------------------------------------------------------------------------
// 1) f32 -> bf16 weight conversion
// ---------------------------------------------------------------------------
__global__ void k_cvt_bf16(const float* __restrict__ src, unsigned short* __restrict__ dst, int n) {
    int id = blockIdx.x * blockDim.x + threadIdx.x;
    if (id < n) dst[id] = f2bf(src[id]);
}

// ---------------------------------------------------------------------------
// 2) rk = aux_keys @ Wr.T   [NSLOT, RDIM]
// ---------------------------------------------------------------------------
__global__ void k_rk(const float* __restrict__ aux_keys, const float* __restrict__ Wr,
                     float* __restrict__ rk) {
    int id = blockIdx.x * blockDim.x + threadIdx.x;
    if (id >= NSLOT * RDIM) return;
    int n = id / RDIM, r = id % RDIM;
    float s = 0.0f;
    const float* a = aux_keys + (size_t)n * BOT;
    const float* w = Wr + (size_t)r * BOT;
    for (int b = 0; b < BOT; ++b) s = fmaf(a[b], w[b], s);
    rk[id] = s;
}

// ---------------------------------------------------------------------------
// 3) AVd = aux_values @ Wvd.T   [NSLOT, VB]  (value bottleneck pre-applied)
// ---------------------------------------------------------------------------
__global__ void k_avd(const float* __restrict__ aux_values, const float* __restrict__ Wvd,
                      float* __restrict__ avd) {
    int id = blockIdx.x * blockDim.x + threadIdx.x;
    if (id >= NSLOT * VB) return;
    int n = id / VB, v = id % VB;
    const float4* a = (const float4*)(aux_values + (size_t)n * HID);
    const float4* w = (const float4*)(Wvd + (size_t)v * HID);
    float s = 0.0f;
    for (int i = 0; i < HID / 4; ++i) {
        float4 av = a[i], wv = w[i];
        s = fmaf(av.x, wv.x, s); s = fmaf(av.y, wv.y, s);
        s = fmaf(av.z, wv.z, s); s = fmaf(av.w, wv.w, s);
    }
    avd[id] = s;
}

// ---------------------------------------------------------------------------
// 4) learned[t] += sum_j gelu((h@Wu1.T)[t,j] + bu1[j]) * Wu2[j]
//    WMMA bf16 GEMM. Block = 64 tokens (4 M-tiles in LDS) x 256-col j-slice.
//    grid = (NTOK/64, 1024/256). Each wave owns 2 j-tiles; every B fragment
//    feeds 4 WMMAs (4x M-reuse -> 4x less B traffic from L2).
// ---------------------------------------------------------------------------
__global__ void __launch_bounds__(256)
k_uncert(const float* __restrict__ h, const unsigned short* __restrict__ Wu1b,
         const float* __restrict__ bu1, const float* __restrict__ Wu2,
         float* __restrict__ learned) {
    __shared__ __align__(16) unsigned short As[64 * 32];   // 64 tokens x 32 k (bf16)
    const int t0   = blockIdx.x * 64;
    const int jg   = blockIdx.y;          // j-group: 256 columns
    const int wave = threadIdx.x >> 5;
    const int lane = threadIdx.x & 31;
    const int nn   = lane & 15;           // column / row-in-tile
    const int gg   = lane >> 4;           // lane half
    const int jt0  = jg * 16 + wave * 2;  // this wave's first j-tile

    v8f zero = {0.f, 0.f, 0.f, 0.f, 0.f, 0.f, 0.f, 0.f};
    v8f acc[2][4];
#pragma unroll
    for (int jt = 0; jt < 2; ++jt)
#pragma unroll
        for (int mt = 0; mt < 4; ++mt) acc[jt][mt] = zero;

    for (int k0 = 0; k0 < HID; k0 += 32) {
        // stage A-tile (64 tokens x 32 k) as bf16 in LDS: 8 elems / thread
        {
            int e = threadIdx.x << 3;       // 2048 elements / 256 threads
            int r = e >> 5, c0 = e & 31;
            const float4* hp = (const float4*)(h + (size_t)(t0 + r) * HID + k0 + c0);
            float4 a0 = hp[0], a1 = hp[1];
            unsigned short* dst = &As[r * 32 + c0];
            dst[0] = f2bf(a0.x); dst[1] = f2bf(a0.y); dst[2] = f2bf(a0.z); dst[3] = f2bf(a0.w);
            dst[4] = f2bf(a1.x); dst[5] = f2bf(a1.y); dst[6] = f2bf(a1.z); dst[7] = f2bf(a1.w);
        }
        __syncthreads();

        Frag16 a[4];
#pragma unroll
        for (int mt = 0; mt < 4; ++mt) {
            int row = mt * 16 + nn;
            a[mt].q[0] = *(const uint4*)&As[row * 32 + gg * 8];
            a[mt].q[1] = *(const uint4*)&As[row * 32 + gg * 8 + 16];
        }

#pragma unroll
        for (int jt = 0; jt < 2; ++jt) {
            int j0 = (jt0 + jt) << 4;
            const unsigned short* bp = Wu1b + (size_t)(j0 + nn) * HID + k0 + gg * 8;
            Frag16 b;
            b.q[0] = *(const uint4*)bp;
            b.q[1] = *(const uint4*)(bp + 16);
#pragma unroll
            for (int mt = 0; mt < 4; ++mt)
                acc[jt][mt] = __builtin_amdgcn_wmma_f32_16x16x32_bf16(
                    false, a[mt].v, false, b.v, (short)0, acc[jt][mt], false, false);
        }
        __syncthreads();
    }

    // epilogue: bias + gelu, scale by Wu2, fold over j (N dim = lanes)
    float tokacc[4][8];
#pragma unroll
    for (int mt = 0; mt < 4; ++mt)
#pragma unroll
        for (int r = 0; r < 8; ++r) tokacc[mt][r] = 0.0f;

#pragma unroll
    for (int jt = 0; jt < 2; ++jt) {
        int j = ((jt0 + jt) << 4) + nn;
        float bias = bu1[j];
        float u2   = Wu2[j];
#pragma unroll
        for (int mt = 0; mt < 4; ++mt)
#pragma unroll
            for (int r = 0; r < 8; ++r)
                tokacc[mt][r] = fmaf(gelu_tanh(acc[jt][mt][r] + bias), u2, tokacc[mt][r]);
    }
    // reduce over the 16 lanes of each half (columns n = 0..15)
#pragma unroll
    for (int mt = 0; mt < 4; ++mt)
#pragma unroll
        for (int r = 0; r < 8; ++r) {
            float v = tokacc[mt][r];
            v += __shfl_xor(v, 1);
            v += __shfl_xor(v, 2);
            v += __shfl_xor(v, 4);
            v += __shfl_xor(v, 8);
            tokacc[mt][r] = v;
        }
    if (nn == 0) {  // lanes 0 and 16
#pragma unroll
        for (int mt = 0; mt < 4; ++mt)
#pragma unroll
            for (int r = 0; r < 8; ++r)
                atomicAdd(&learned[t0 + mt * 16 + r + 8 * gg], tokacc[mt][r]);
    }
}

// ---------------------------------------------------------------------------
// 5) query = h @ Wq.T   [NTOK, BOT]   (WMMA bf16, 4 waves = 4 n-tiles)
// ---------------------------------------------------------------------------
__global__ void __launch_bounds__(128)
k_query(const float* __restrict__ h, const unsigned short* __restrict__ Wqb,
        float* __restrict__ query) {
    __shared__ __align__(16) unsigned short As[16 * 32];
    const int t0   = blockIdx.x * 16;
    const int wave = threadIdx.x >> 5;
    const int lane = threadIdx.x & 31;
    const int nn   = lane & 15;
    const int gg   = lane >> 4;

    v8f acc = {0.f, 0.f, 0.f, 0.f, 0.f, 0.f, 0.f, 0.f};

    for (int k0 = 0; k0 < HID; k0 += 32) {
        {
            int e = threadIdx.x << 2;          // 512 elements / 128 threads
            int r = e >> 5, c = e & 31;
            const float4 hv = *(const float4*)(h + (size_t)(t0 + r) * HID + k0 + c);
            As[r * 32 + c]     = f2bf(hv.x);
            As[r * 32 + c + 1] = f2bf(hv.y);
            As[r * 32 + c + 2] = f2bf(hv.z);
            As[r * 32 + c + 3] = f2bf(hv.w);
        }
        __syncthreads();

        Frag16 a;
        a.q[0] = *(const uint4*)&As[nn * 32 + gg * 8];
        a.q[1] = *(const uint4*)&As[nn * 32 + gg * 8 + 16];

        const unsigned short* bp = Wqb + (size_t)(wave * 16 + nn) * HID + k0 + gg * 8;
        Frag16 b;
        b.q[0] = *(const uint4*)bp;
        b.q[1] = *(const uint4*)(bp + 16);
        acc = __builtin_amdgcn_wmma_f32_16x16x32_bf16(
            false, a.v, false, b.v, (short)0, acc, false, false);
        __syncthreads();
    }
#pragma unroll
    for (int r = 0; r < 8; ++r) {
        int m = r + 8 * gg;
        query[(size_t)(t0 + m) * BOT + wave * 16 + nn] = acc[r];
    }
}

// ---------------------------------------------------------------------------
// 6) per-token variance -> log1p, + global sum  (one wave per token)
// ---------------------------------------------------------------------------
__global__ void __launch_bounds__(256)
k_var(const float* __restrict__ h, float* __restrict__ logvar, float* __restrict__ lv_sum) {
    const int wave = threadIdx.x >> 5;
    const int lane = threadIdx.x & 31;
    const int t = blockIdx.x * 8 + wave;
    const float4* hp = (const float4*)(h + (size_t)t * HID);
    float s = 0.0f, s2 = 0.0f;
    for (int i = lane; i < HID / 4; i += 32) {
        float4 v = hp[i];
        s  += v.x + v.y + v.z + v.w;
        s2  = fmaf(v.x, v.x, s2); s2 = fmaf(v.y, v.y, s2);
        s2  = fmaf(v.z, v.z, s2); s2 = fmaf(v.w, v.w, s2);
    }
#pragma unroll
    for (int off = 16; off > 0; off >>= 1) {
        s  += __shfl_xor(s, off);
        s2 += __shfl_xor(s2, off);
    }
    if (lane == 0) {
        float mean = s * (1.0f / HID);
        float var  = s2 * (1.0f / HID) - mean * mean;
        float lv   = log1pf(var);
        logvar[t] = lv;
        atomicAdd(lv_sum, lv);
    }
}

// ---------------------------------------------------------------------------
// 7) router + top-k + softmax + z[t,0:8] = sum_k w_k * AVd[idx_k]
//    one wave per token, 8 tokens per block
// ---------------------------------------------------------------------------
__global__ void __launch_bounds__(256)
k_route(const float* __restrict__ query, const float* __restrict__ Wr,
        const float* __restrict__ rk, const float* __restrict__ log_rel,
        const float* __restrict__ aux_keys, const float* __restrict__ avd,
        float* __restrict__ z) {
    __shared__ float sc[8][NSLOT];
    __shared__ float qs[8][BOT];
    __shared__ float rqs[8][RDIM];
    const int wave = threadIdx.x >> 5;
    const int lane = threadIdx.x & 31;
    const int t = blockIdx.x * 8 + wave;

    // load query row
    qs[wave][lane * 2]     = query[(size_t)t * BOT + lane * 2];
    qs[wave][lane * 2 + 1] = query[(size_t)t * BOT + lane * 2 + 1];
    __syncthreads();

    // rq = q @ Wr.T
    for (int r = lane; r < RDIM; r += 32) {
        float s = 0.0f;
        const float* w = Wr + (size_t)r * BOT;
        for (int b = 0; b < BOT; ++b) s = fmaf(qs[wave][b], w[b], s);
        rqs[wave][r] = s;
    }
    __syncthreads();

    // router scores over all slots
    const float inv_sqrt_r = 0.14433756729740643f;  // 1/sqrt(48)
    for (int n = lane; n < NSLOT; n += 32) {
        float s = 0.0f;
        const float* kk = rk + (size_t)n * RDIM;
        for (int r = 0; r < RDIM; ++r) s = fmaf(rqs[wave][r], kk[r], s);
        sc[wave][n] = s * inv_sqrt_r + log_rel[n];
    }
    __syncthreads();

    // iterative top-8 (wave argmax over LDS scores)
    int   topi[KSEL];
    for (int it = 0; it < KSEL; ++it) {
        float bv = -3.0e38f;
        int   bi = NSLOT;
        for (int n = lane; n < NSLOT; n += 32) {
            float s = sc[wave][n];
            if (s > bv || (s == bv && n < bi)) { bv = s; bi = n; }
        }
#pragma unroll
        for (int off = 16; off > 0; off >>= 1) {
            float ov = __shfl_xor(bv, off);
            int   oi = __shfl_xor(bi, off);
            if (ov > bv || (ov == bv && oi < bi)) { bv = ov; bi = oi; }
        }
        topi[it] = bi;
        if (lane == 0) sc[wave][bi] = -3.0e38f;
        __syncthreads();
    }

    // attention over the 8 selected slots (full-precision dot over BOT)
    float attn[KSEL];
#pragma unroll
    for (int k = 0; k < KSEL; ++k) {
        int nk = topi[k];
        const float* ak = aux_keys + (size_t)nk * BOT;
        float p = fmaf(qs[wave][lane * 2],     ak[lane * 2],     0.0f);
        p       = fmaf(qs[wave][lane * 2 + 1], ak[lane * 2 + 1], p);
#pragma unroll
        for (int off = 16; off > 0; off >>= 1) p += __shfl_xor(p, off);
        attn[k] = p * 0.125f + log_rel[nk];   // /sqrt(64)
    }
    // softmax (replicated per lane)
    float m = attn[0];
#pragma unroll
    for (int k = 1; k < KSEL; ++k) m = fmaxf(m, attn[k]);
    float denom = 0.0f;
    float wk[KSEL];
#pragma unroll
    for (int k = 0; k < KSEL; ++k) { wk[k] = __expf(attn[k] - m); denom += wk[k]; }
    float inv = 1.0f / denom;

    // z[t, v] = sum_k w_k * AVd[idx_k, v]
    if (lane < VB) {
        float zv = 0.0f;
#pragma unroll
        for (int k = 0; k < KSEL; ++k)
            zv = fmaf(wk[k] * inv, avd[(size_t)topi[k] * VB + lane], zv);
        z[(size_t)t * VB + lane] = zv;
    }
}

// ---------------------------------------------------------------------------
// 8) gate[t] from variance stats + learned MLP head
// ---------------------------------------------------------------------------
__global__ void k_gate(const float* __restrict__ logvar, const float* __restrict__ lv_sum,
                       const float* __restrict__ learned, const float* __restrict__ bu2,
                       const float* __restrict__ gw1, const float* __restrict__ gb,
                       float* __restrict__ gate) {
    int t = blockIdx.x * blockDim.x + threadIdx.x;
    if (t >= NTOK) return;
    float mean_lv  = lv_sum[0] * (1.0f / NTOK);
    float norm_var = logvar[t] / (mean_lv + 1e-6f);
    float lrn      = learned[t] + bu2[0];
    float sig      = 1.0f / (1.0f + __expf(-lrn));
    float u        = fminf(fmaxf(norm_var * 0.5f + sig * 2.5f, 0.0f), 5.0f);
    float un       = fminf(fmaxf((u - 0.5f) * (1.0f / 1.5f), 0.0f), 1.0f);
    float g        = 1.0f / (1.0f + __expf(-(gw1[0] * un + gb[0])));
    gate[t] = (g < 0.05f) ? 0.0f : g;
}

// ---------------------------------------------------------------------------
// 9) out = h + gate[t] * (z[t] @ Wvu.T)   (rank-8 update, fused)
// ---------------------------------------------------------------------------
__global__ void __launch_bounds__(256)
k_final(const float* __restrict__ h, const float* __restrict__ z,
        const float* __restrict__ gate, const float* __restrict__ Wvu,
        float* __restrict__ out) {
    int id = blockIdx.x * blockDim.x + threadIdx.x;   // NTOK * HID/4 threads
    int t  = id >> 10;            // HID/4 = 1024
    int c  = id & 1023;
    int h0 = c << 2;
    float g = gate[t];
    const float* zt = z + (size_t)t * VB;
    float zv[VB];
#pragma unroll
    for (int v = 0; v < VB; ++v) zv[v] = zt[v];
    float4 hv = *(const float4*)(h + (size_t)t * HID + h0);
    float4 ov;
    float* po = &ov.x;
    const float* ph = &hv.x;
#pragma unroll
    for (int i = 0; i < 4; ++i) {
        const float4* wr = (const float4*)(Wvu + (size_t)(h0 + i) * VB);
        float4 w0 = wr[0], w1 = wr[1];
        float d = 0.0f;
        d = fmaf(zv[0], w0.x, d); d = fmaf(zv[1], w0.y, d);
        d = fmaf(zv[2], w0.z, d); d = fmaf(zv[3], w0.w, d);
        d = fmaf(zv[4], w1.x, d); d = fmaf(zv[5], w1.y, d);
        d = fmaf(zv[6], w1.z, d); d = fmaf(zv[7], w1.w, d);
        po[i] = ph[i] + g * d;
    }
    *(float4*)(out + (size_t)t * HID + h0) = ov;
}

// ---------------------------------------------------------------------------
extern "C" void kernel_launch(void* const* d_in, const int* in_sizes, int n_in,
                              void* d_out, int out_size, void* d_ws, size_t ws_size,
                              hipStream_t stream) {
    const float* h        = (const float*)d_in[0];
    const float* Wq       = (const float*)d_in[1];
    const float* Wr       = (const float*)d_in[2];
    const float* aux_keys = (const float*)d_in[3];
    const float* aux_vals = (const float*)d_in[4];
    const float* Wvd      = (const float*)d_in[5];
    const float* Wvu      = (const float*)d_in[6];
    const float* Wu1      = (const float*)d_in[7];
    const float* bu1      = (const float*)d_in[8];
    const float* Wu2      = (const float*)d_in[9];
    const float* bu2      = (const float*)d_in[10];
    const float* gw1      = (const float*)d_in[11];
    const float* gb       = (const float*)d_in[12];
    const float* log_rel  = (const float*)d_in[13];
    float* out = (float*)d_out;

    char* ws = (char*)d_ws;
    size_t off = 0;
    auto alloc = [&](size_t bytes) { char* p = ws + off; off = (off + bytes + 255) & ~(size_t)255; return p; };
    unsigned short* Wu1b   = (unsigned short*)alloc((size_t)H4 * HID * 2);
    unsigned short* Wqb    = (unsigned short*)alloc((size_t)BOT * HID * 2);
    float* query   = (float*)alloc((size_t)NTOK * BOT * 4);
    float* rk      = (float*)alloc((size_t)NSLOT * RDIM * 4);
    float* avd     = (float*)alloc((size_t)NSLOT * VB * 4);
    float* z       = (float*)alloc((size_t)NTOK * VB * 4);
    float* learned = (float*)alloc((size_t)NTOK * 4);
    float* logvar  = (float*)alloc((size_t)NTOK * 4);
    float* gate    = (float*)alloc((size_t)NTOK * 4);
    float* lv_sum  = (float*)alloc(256);
    (void)ws_size; (void)in_sizes; (void)n_in; (void)out_size;

    k_zero<<<(NTOK + 255) / 256, 256, 0, stream>>>(learned, lv_sum);
    k_cvt_bf16<<<(H4 * HID + 255) / 256, 256, 0, stream>>>(Wu1, Wu1b, H4 * HID);
    k_cvt_bf16<<<(BOT * HID + 255) / 256, 256, 0, stream>>>(Wq, Wqb, BOT * HID);
    k_rk<<<(NSLOT * RDIM + 255) / 256, 256, 0, stream>>>(aux_keys, Wr, rk);
    k_avd<<<(NSLOT * VB + 255) / 256, 256, 0, stream>>>(aux_vals, Wvd, avd);

    dim3 ugrid(NTOK / 64, H4 / 256);
    k_uncert<<<ugrid, 256, 0, stream>>>(h, Wu1b, bu1, Wu2, learned);
    k_query <<<NTOK / 16, 128, 0, stream>>>(h, Wqb, query);
    k_var   <<<NTOK / 8,  256, 0, stream>>>(h, logvar, lv_sum);
    k_route <<<NTOK / 8,  256, 0, stream>>>(query, Wr, rk, log_rel, aux_keys, avd, z);
    k_gate  <<<NTOK / 256, 256, 0, stream>>>(logvar, lv_sum, learned, bu2, gw1, gb, gate);
    k_final <<<(NTOK * (HID / 4)) / 256, 256, 0, stream>>>(h, z, gate, Wvu, out);
}
